// GATModel_30949534335634
// MI455X (gfx1250) — matrix operand order repaired and checked by reading, not compiled, not run on hardware
//
#include <hip/hip_runtime.h>
#include <math.h>

// ---------------------------------------------------------------------------
// 2-layer GAT for gfx1250 (MI455X).
//   L1: h = x@W1 (bf16 WMMA, f32 acc) -> per-head attn -> segment softmax ->
//       scatter-add -> +b1, ReLU
//   L2: z = h1@W2 (wave GEMV) -> 1-head attn -> segment softmax -> scatter ->
//       +b2
// x / W1^T are pre-converted to bf16 once; GEMM blocks cover 128x128 so each
// wave issues 32 WMMAs per W1 LDS fill. Edge passes recompute exp() (all
// per-node arrays are L2-resident; only the edge list streams from HBM).
// ---------------------------------------------------------------------------

typedef __attribute__((ext_vector_type(16))) __bf16       v16bf;
typedef __attribute__((ext_vector_type(8)))  float        v8f;
typedef __attribute__((ext_vector_type(4)))  unsigned int v4u;

union BF16Frag { v16bf v; v4u q[2]; };

static __device__ __forceinline__ unsigned short f2bf(float f) {
    unsigned u = __float_as_uint(f);
    unsigned r = (u + 0x7FFFu + ((u >> 16) & 1u)) >> 16;   // RNE
    return (unsigned short)r;
}

// order-preserving float -> uint map (so atomicMax(u32) == float max, init 0)
static __device__ __forceinline__ unsigned fmap(float f) {
    unsigned u = __float_as_uint(f);
    return (u & 0x80000000u) ? ~u : (u | 0x80000000u);
}
static __device__ __forceinline__ float funmap(unsigned u) {
    unsigned v = (u & 0x80000000u) ? (u & 0x7FFFFFFFu) : ~u;
    return __uint_as_float(v);
}

static __device__ __forceinline__ float leaky(float v) {
    return v > 0.f ? v : 0.2f * v;
}

static __device__ __forceinline__ void edge_sd(const int* __restrict__ ei,
                                               int E, int e, int& s, int& d) {
    if (e < E) { s = ei[e]; d = ei[E + e]; }
    else       { s = e - E; d = e - E; }          // appended self-loop
}

// ---------------------------------------------------------------------------
// One-shot bf16 conversions.
// ---------------------------------------------------------------------------
__global__ void cvt_bf16_kernel(const float* __restrict__ src,
                                unsigned short* __restrict__ dst, int total4) {
    int i = blockIdx.x * blockDim.x + threadIdx.x;
    if (i >= total4) return;
    const float4 v = ((const float4*)src)[i];
    union { unsigned short s[4]; unsigned long long ll; } p;
    p.s[0] = f2bf(v.x); p.s[1] = f2bf(v.y); p.s[2] = f2bf(v.z); p.s[3] = f2bf(v.w);
    ((unsigned long long*)dst)[i] = p.ll;
}

__global__ void cvt_w1t_kernel(const float* __restrict__ W1,
                               unsigned short* __restrict__ w1t) {
    int i = blockIdx.x * blockDim.x + threadIdx.x;   // i = k*128 + c
    if (i >= 128 * 128) return;
    int k = i >> 7, c = i & 127;
    w1t[c * 128 + k] = f2bf(W1[i]);
}

// ---------------------------------------------------------------------------
// K1: hlin[N,128] = x[N,128] @ W1[128,128]   (bf16 WMMA, f32 accumulate)
// block = 256 thr = 8 waves; block tile 128 rows x 128 cols;
// wave owns one 16-col stripe and loops 8 row-subtiles -> 32 WMMAs/wave.
// ---------------------------------------------------------------------------
__global__ __launch_bounds__(256) void gemm1_wmma_kernel(
        const unsigned short* __restrict__ xb,   // bf16 x [N,128]
        const unsigned short* __restrict__ w1t,  // bf16 W1^T [128,128]
        float* __restrict__ hlin, int n) {
    __shared__ __align__(16) unsigned short xs[128 * 128];  // 32KB
    __shared__ __align__(16) unsigned short wt[128 * 128];  // 32KB

    const int tid     = threadIdx.x;
    const int rowbase = blockIdx.x * 128;

    // raw b128 copies into LDS (already bf16)
    for (int i = tid * 8; i < 128 * 128; i += 256 * 8)
        *(v4u*)&wt[i] = *(const v4u*)&w1t[i];
    for (int i = tid * 8; i < 128 * 128; i += 256 * 8) {
        int r  = rowbase + (i >> 7);
        int rr = r < n ? r : (n - 1);
        *(v4u*)&xs[i] = *(const v4u*)&xb[(size_t)rr * 128 + (i & 127)];
    }
    __syncthreads();

    const int wave    = tid >> 5;        // 0..7  -> 16-col stripe
    const int lane    = tid & 31;
    const int colbase = wave * 16;
    const int mrow    = lane & 15;
    const int hi      = lane >> 4;       // half-wave select

    for (int sub = 0; sub < 8; ++sub) {
        const int r0 = sub * 16;
        v8f acc = {};
        #pragma unroll
        for (int kk = 0; kk < 128; kk += 32) {
            // A frag: row mrow, K segs {0..7,16..23} (hi=0) / {8..15,24..31}
            BF16Frag a, b;
            const unsigned short* ap = &xs[(r0 + mrow) * 128 + kk + hi * 8];
            a.q[0] = *(const v4u*)(ap);
            a.q[1] = *(const v4u*)(ap + 16);
            // B frag: col colbase+mrow, K segs {0..7,16..23} (hi=0) / {8..15,24..31}
            const unsigned short* bp = &wt[(colbase + mrow) * 128 + kk + hi * 8];
            b.q[0] = *(const v4u*)(bp);
            b.q[1] = *(const v4u*)(bp + 16);
            acc = __builtin_amdgcn_wmma_f32_16x16x32_bf16(
                      false, a.v, false, b.v, (short)0, acc, false, false);
        }
        // C/D layout: VGPR r -> row (r + 8*hi), col (lane%16)
        const int rowt = rowbase + r0;
        if (rowt + 16 <= n) {            // fast path: whole subtile in range
            float* out = hlin + (size_t)(rowt + 8 * hi) * 128 + colbase + mrow;
            #pragma unroll
            for (int r = 0; r < 8; ++r) out[(size_t)r * 128] = acc[r];
        } else {
            #pragma unroll
            for (int r = 0; r < 8; ++r) {
                int row = rowt + r + 8 * hi;
                if (row < n) hlin[(size_t)row * 128 + colbase + mrow] = acc[r];
            }
        }
    }
}

// ---------------------------------------------------------------------------
// K2: per-(node,head) attention logits  a_s = <h, att_src>, a_d = <h, att_dst>
// ---------------------------------------------------------------------------
__global__ void dot_heads_kernel(const float* __restrict__ hlin,
                                 const float* __restrict__ att_src,
                                 const float* __restrict__ att_dst,
                                 float* __restrict__ as_, float* __restrict__ ad_,
                                 int n, int H, int C) {
    int idx = blockIdx.x * blockDim.x + threadIdx.x;
    if (idx >= n * H) return;
    int node = idx / H, h = idx - node * H;
    const float* row = hlin + (size_t)node * H * C + h * C;
    float s = 0.f, d = 0.f;
    for (int c = 0; c < C; ++c) {
        float v = row[c];
        s += v * att_src[h * C + c];
        d += v * att_dst[h * C + c];
    }
    as_[idx] = s; ad_[idx] = d;
}

// ---------------------------------------------------------------------------
// K3/K4/K5: edge-parallel segment softmax + scatter (H, C compile-time)
// ---------------------------------------------------------------------------
template <int H>
__global__ void edge_max_kernel(const int* __restrict__ ei, int E, int ET,
                                const float* __restrict__ as_,
                                const float* __restrict__ ad_,
                                unsigned* __restrict__ m) {
    int idx = blockIdx.x * blockDim.x + threadIdx.x;
    if (idx >= ET * H) return;
    int e = idx / H, h = idx - e * H;
    int s, d; edge_sd(ei, E, e, s, d);
    float v = leaky(as_[s * H + h] + ad_[d * H + h]);
    atomicMax(&m[d * H + h], fmap(v));
}

template <int H>
__global__ void edge_sum_kernel(const int* __restrict__ ei, int E, int ET,
                                const float* __restrict__ as_,
                                const float* __restrict__ ad_,
                                const unsigned* __restrict__ m,
                                float* __restrict__ ssum) {
    int idx = blockIdx.x * blockDim.x + threadIdx.x;
    if (idx >= ET * H) return;
    int e = idx / H, h = idx - e * H;
    int s, d; edge_sd(ei, E, e, s, d);
    float v = leaky(as_[s * H + h] + ad_[d * H + h]);
    float p = __expf(v - funmap(m[d * H + h]));
    atomicAdd(&ssum[d * H + h], p);
}

template <int H, int C>
__global__ void edge_agg_kernel(const int* __restrict__ ei, int E, int ET,
                                const float* __restrict__ as_,
                                const float* __restrict__ ad_,
                                const unsigned* __restrict__ m,
                                const float* __restrict__ ssum,
                                const float* __restrict__ feat,
                                float* __restrict__ agg) {
    int idx = blockIdx.x * blockDim.x + threadIdx.x;
    if (idx >= ET * H) return;
    int e = idx / H, h = idx - e * H;
    int s, d; edge_sd(ei, E, e, s, d);
    float v = leaky(as_[s * H + h] + ad_[d * H + h]);
    float alpha = __expf(v - funmap(m[d * H + h])) / ssum[d * H + h];
    constexpr int F = H * C;
    const float* fs = feat + (size_t)s * F + h * C;
    float*       od = agg  + (size_t)d * F + h * C;
    if constexpr ((C & 3) == 0) {
        #pragma unroll
        for (int c4 = 0; c4 < C / 4; ++c4) {         // b128 gathers
            float4 v4 = ((const float4*)fs)[c4];
            atomicAdd(&od[c4 * 4 + 0], v4.x * alpha);
            atomicAdd(&od[c4 * 4 + 1], v4.y * alpha);
            atomicAdd(&od[c4 * 4 + 2], v4.z * alpha);
            atomicAdd(&od[c4 * 4 + 3], v4.w * alpha);
        }
    } else {
        #pragma unroll
        for (int c = 0; c < C; ++c)
            atomicAdd(&od[c], fs[c] * alpha);
    }
}

// ---------------------------------------------------------------------------
// K6: h1 = relu(agg + b1)  (in place over agg)
// ---------------------------------------------------------------------------
__global__ void bias_relu_kernel(float* __restrict__ h1,
                                 const float* __restrict__ b, int total) {
    int i = blockIdx.x * blockDim.x + threadIdx.x;
    if (i >= total) return;
    float v = h1[i] + b[i & 127];
    h1[i] = v > 0.f ? v : 0.f;
}

// ---------------------------------------------------------------------------
// K7: layer-2 GEMV z = h1 @ W2 (+ attn logits), one wave per node
// ---------------------------------------------------------------------------
__global__ __launch_bounds__(256) void gemv2_kernel(
        const float* __restrict__ h1, const float* __restrict__ W2,
        const float* __restrict__ att_src2, const float* __restrict__ att_dst2,
        float* __restrict__ z, float* __restrict__ as2, float* __restrict__ ad2,
        int n) {
    int gtid = blockIdx.x * blockDim.x + threadIdx.x;
    int wid  = gtid >> 5;
    int lane = threadIdx.x & 31;
    if (wid >= n) return;
    const float* row = h1 + (size_t)wid * 128;
    float sum = 0.f;
    #pragma unroll
    for (int c = 0; c < 128; c += 32) sum += row[c + lane] * W2[c + lane];
    #pragma unroll
    for (int off = 16; off > 0; off >>= 1) sum += __shfl_down(sum, off, 32);
    if (lane == 0) {
        z[wid]   = sum;
        as2[wid] = sum * att_src2[0];
        ad2[wid] = sum * att_dst2[0];
    }
}

__global__ void bias_out_kernel(const float* __restrict__ agg2,
                                const float* __restrict__ b2,
                                float* __restrict__ out, int n) {
    int i = blockIdx.x * blockDim.x + threadIdx.x;
    if (i >= n) return;
    out[i] = agg2[i] + b2[0];
}

// ---------------------------------------------------------------------------
extern "C" void kernel_launch(void* const* d_in, const int* in_sizes, int n_in,
                              void* d_out, int out_size, void* d_ws, size_t ws_size,
                              hipStream_t stream) {
    const float* x        = (const float*)d_in[0];
    const float* W1       = (const float*)d_in[1];
    const float* att_src1 = (const float*)d_in[2];
    const float* att_dst1 = (const float*)d_in[3];
    const float* b1       = (const float*)d_in[4];
    const float* W2       = (const float*)d_in[5];
    const float* att_src2 = (const float*)d_in[6];
    const float* att_dst2 = (const float*)d_in[7];
    const float* b2       = (const float*)d_in[8];
    const int*   ei       = (const int*)d_in[9];

    const int N  = in_sizes[0] / 128;
    const int E  = in_sizes[9] / 2;
    const int ET = E + N;                       // with self-loops
    const int H = 4, C = 32;

    // ---- workspace carve-out ----
    char*  ws  = (char*)d_ws;
    size_t off = 0;
    auto alloc = [&](size_t bytes) -> void* {
        void* p = ws + off;
        off = (off + bytes + 255) & ~(size_t)255;
        return p;
    };
    unsigned short* xb   = (unsigned short*)alloc((size_t)N * 128 * 2);
    unsigned short* w1t  = (unsigned short*)alloc((size_t)128 * 128 * 2);
    float*    hlin = (float*)   alloc((size_t)N * 128 * 4);
    float*    h1   = (float*)   alloc((size_t)N * 128 * 4);   // agg1 then h1
    float*    as1  = (float*)   alloc((size_t)N * H * 4);
    float*    ad1  = (float*)   alloc((size_t)N * H * 4);
    unsigned* m1   = (unsigned*)alloc((size_t)N * H * 4);
    float*    s1   = (float*)   alloc((size_t)N * H * 4);
    float*    z    = (float*)   alloc((size_t)N * 4);
    float*    as2  = (float*)   alloc((size_t)N * 4);
    float*    ad2  = (float*)   alloc((size_t)N * 4);
    unsigned* m2   = (unsigned*)alloc((size_t)N * 4);
    float*    s2   = (float*)   alloc((size_t)N * 4);
    float*    agg2 = (float*)   alloc((size_t)N * 4);
    (void)ws_size; (void)n_in; (void)out_size;

    // ---- zero accumulators (0u == fmap(-inf) sentinel for m) ----
    hipMemsetAsync(h1,   0, (size_t)N * 128 * 4, stream);
    hipMemsetAsync(m1,   0, (size_t)N * H * 4,   stream);
    hipMemsetAsync(s1,   0, (size_t)N * H * 4,   stream);
    hipMemsetAsync(m2,   0, (size_t)N * 4,       stream);
    hipMemsetAsync(s2,   0, (size_t)N * 4,       stream);
    hipMemsetAsync(agg2, 0, (size_t)N * 4,       stream);

    const int TB = 256;
    const int edge_blocks_h = (ET * H + TB - 1) / TB;
    const int edge_blocks_1 = (ET + TB - 1) / TB;

    // ---- bf16 staging ----
    cvt_bf16_kernel<<<((N * 128 / 4) + TB - 1) / TB, TB, 0, stream>>>(
        x, xb, N * 128 / 4);
    cvt_w1t_kernel<<<(128 * 128 + TB - 1) / TB, TB, 0, stream>>>(W1, w1t);

    // ---- layer 1 ----
    gemm1_wmma_kernel<<<(N + 127) / 128, TB, 0, stream>>>(xb, w1t, hlin, N);
    dot_heads_kernel<<<(N * H + TB - 1) / TB, TB, 0, stream>>>(
        hlin, att_src1, att_dst1, as1, ad1, N, H, C);
    edge_max_kernel<4><<<edge_blocks_h, TB, 0, stream>>>(ei, E, ET, as1, ad1, m1);
    edge_sum_kernel<4><<<edge_blocks_h, TB, 0, stream>>>(ei, E, ET, as1, ad1, m1, s1);
    edge_agg_kernel<4, 32><<<edge_blocks_h, TB, 0, stream>>>(
        ei, E, ET, as1, ad1, m1, s1, hlin, h1);
    bias_relu_kernel<<<((size_t)N * 128 + TB - 1) / TB, TB, 0, stream>>>(h1, b1, N * 128);

    // ---- layer 2 (H=1, C=1) ----
    gemv2_kernel<<<(N * 32 + TB - 1) / TB, TB, 0, stream>>>(
        h1, W2, att_src2, att_dst2, z, as2, ad2, N);
    edge_max_kernel<1><<<edge_blocks_1, TB, 0, stream>>>(ei, E, ET, as2, ad2, m2);
    edge_sum_kernel<1><<<edge_blocks_1, TB, 0, stream>>>(ei, E, ET, as2, ad2, m2, s2);
    edge_agg_kernel<1, 1><<<edge_blocks_1, TB, 0, stream>>>(
        ei, E, ET, as2, ad2, m2, s2, z, agg2);
    bias_out_kernel<<<(N + TB - 1) / TB, TB, 0, stream>>>(agg2, b2, (float*)d_out, N);
}
